// Convpass_swin_hypernet_diag_22359599743440
// MI455X (gfx1250) — matrix operand order repaired and checked by reading, not compiled, not run on hardware
//
#include <hip/hip_runtime.h>

// ---------------------------------------------------------------------------
// Convpass (swin, hypernet, diag) forward for MI455X (gfx1250, wave32, WMMA).
// Pipeline: hypernet GEMV -> down-proj GEMM(+gelu) -> 3x3 conv (9 GEMMs)
//           -> scale+gelu -> up-proj GEMM.  All big GEMMs via
//           v_wmma_f32_16x16x32_f16 (f16 inputs, f32 accumulate).
// Memory-bound (~206MB mandatory HBM traffic). Intermediates live in
// L2-resident workspace.  Each wave handles TWO 16-pixel M-tiles so every
// B-fragment load (L2) feeds two WMMAs.  quickGELU uses v_rcp_f32, not the
// IEEE divide sequence.
// ---------------------------------------------------------------------------

#define Cc    512
#define DIMc  64
#define Hc    28
#define Wc    28
#define PIX   784          // 28*28
#define TILES_PER_IMG 49   // 784/16
#define NTILES 3136        // 64*49  (16-pixel M-tiles)
#define NJOBS  1568        // 2 M-tiles per wave
#define HYPN  36864        // DIM*DIM*9

typedef _Float16 half16 __attribute__((ext_vector_type(16)));
typedef float    float8 __attribute__((ext_vector_type(8)));

union FragU { half16 h; unsigned u[8]; };

static __device__ inline unsigned pk2(float a, float b) {
  // v_cvt_pk_rtz_f16_f32; builtin returns __fp16x2 -> just take the 32 bits.
  return __builtin_bit_cast(unsigned, __builtin_amdgcn_cvt_pkrtz(a, b));
}

// 16-bit A-matrix 16x32 layout (ISA 7.12.2): lane holds row m=lane&15,
// VGPR v holds K = (v>>2)*16 + (lane>>4)*8 + (v&3)*2  (two consecutive f16).
static __device__ inline int a_koff(int v, int hi) {
  return ((v >> 2) << 4) + (hi << 3) + ((v & 3) << 1);
}

static __device__ inline float quick_gelu(float y) {
  // y * sigmoid(1.702 y) with v_exp_f32 + v_rcp_f32 (no IEEE div chain).
  return y * __builtin_amdgcn_rcpf(1.0f + __expf(-1.702f * y));
}

static __device__ inline float8 wmma_f16(const FragU& a, const FragU& b, float8 c) {
  // 8 args: (neg_a, A, neg_b, B, c_mod, C, reuse_a, reuse_b)
  return __builtin_amdgcn_wmma_f32_16x16x32_f16(false, a.h, false, b.h,
                                                (short)0, c, false, false);
}

static __device__ inline void load_b(FragU& b, const unsigned* __restrict__ base) {
  const uint4* bp = (const uint4*)base;
  *(uint4*)&b.u[0] = bp[0];
  *(uint4*)&b.u[4] = bp[1];
}

#define ZERO8 float8{0.f,0.f,0.f,0.f,0.f,0.f,0.f,0.f}

// ---------------------------------------------------------------------------
// k_prep: hypernet GEMV + pre-swizzle all weights into per-lane B-fragment
// order.  B 32x16 fragment: lane holds col n=lane&15, K = a_koff(v, lane>>4).
// Fragment buffers are indexed [..][lane][v] so GEMM kernels read 32B/lane
// fully coalesced.
//   convw_frag : [t=9][nt=4][kb=2][lane=32][v=8]  dwords (18432)
//   wdown_frag : [nt=4][kb=16][lane=32][v=8]      dwords (16384)
//   wup_frag   : [nt=32][kb=2][lane=32][v=8]      dwords (16384)
// ---------------------------------------------------------------------------
__global__ __launch_bounds__(256)
void k_prep(const float* __restrict__ w_down, const float* __restrict__ w_up,
            const float* __restrict__ layer_emb, const float* __restrict__ w_hyper,
            const float* __restrict__ b_hyper,
            unsigned* __restrict__ convw_frag, unsigned* __restrict__ wdown_frag,
            unsigned* __restrict__ wup_frag) {
  const int tid = blockIdx.x * 256 + threadIdx.x;
  if (tid < 18432) {
    // hypernet GEMV -> conv weight fragment.  conv_w is OIHW [oo][ii][kh][kw].
    const int s = tid;
    const int v = s & 7, lane = (s >> 3) & 31, kb = (s >> 8) & 1;
    const int nt = (s >> 9) & 3, t = s >> 11;            // t in 0..8
    const int oo = nt * 16 + (lane & 15);                // out channel (N)
    const int k0 = kb * 32 + a_koff(v, lane >> 4);       // in channel (K)
    const int kh = t / 3, kw = t % 3;
    const int o0 = ((oo * 64 + k0) * 3 + kh) * 3 + kw;
    const int o1 = ((oo * 64 + k0 + 1) * 3 + kh) * 3 + kw;
    float s0 = b_hyper[o0], s1 = b_hyper[o1];
    for (int e = 0; e < 64; ++e) {
      const float le = layer_emb[e];
      s0 += le * w_hyper[(size_t)e * HYPN + o0];
      s1 += le * w_hyper[(size_t)e * HYPN + o1];
    }
    convw_frag[tid] = pk2(s0, s1);
  } else if (tid < 18432 + 16384) {
    const int s = tid - 18432;
    const int v = s & 7, lane = (s >> 3) & 31, kb = (s >> 8) & 15, nt = (s >> 12) & 3;
    const int c = nt * 16 + (lane & 15);
    const int k = kb * 32 + a_koff(v, lane >> 4);
    wdown_frag[s] = pk2(w_down[k * DIMc + c], w_down[(k + 1) * DIMc + c]);
  } else if (tid < 18432 + 16384 + 16384) {
    const int s = tid - 34816;
    const int v = s & 7, lane = (s >> 3) & 31, kb = (s >> 8) & 1, nt = s >> 9;
    const int c = nt * 16 + (lane & 15);
    const int k = kb * 32 + a_koff(v, lane >> 4);
    wup_frag[s] = pk2(w_up[(size_t)k * Cc + c], w_up[(size_t)(k + 1) * Cc + c]);
  }
}

// ---------------------------------------------------------------------------
// k_down: act = quick_gelu(x @ w_down + b_down), stored f16 [pixel][64].
// One wave per TWO 16-pixel M-tiles (M=32), N=64, K=512 in 16 blocks of 32
// with on-the-fly f32->f16 pack of A.  Each B load feeds two WMMAs.
// ---------------------------------------------------------------------------
__global__ __launch_bounds__(256)
void k_down(const float* __restrict__ x, const float* __restrict__ b_down,
            const unsigned* __restrict__ wdown_frag, _Float16* __restrict__ act) {
  const int lane = threadIdx.x & 31;
  const int wave = threadIdx.x >> 5;
  const int job = blockIdx.x * 8 + wave;         // 0..1567
  const int mt0 = job * 2;
  const int m = lane & 15, hi = lane >> 4;
  const float* xrow0 = x + (size_t)(mt0 * 16 + m) * Cc;
  const float* xrow1 = xrow0 + (size_t)16 * Cc;

  float8 acc[2][4];
  #pragma unroll
  for (int ti = 0; ti < 2; ++ti)
    #pragma unroll
    for (int i = 0; i < 4; ++i) acc[ti][i] = ZERO8;

  for (int kb = 0; kb < 16; ++kb) {
    FragU a0, a1;
    #pragma unroll
    for (int v = 0; v < 8; ++v) {
      const int ko = kb * 32 + a_koff(v, hi);
      const float2 f0 = *(const float2*)(xrow0 + ko);
      const float2 f1 = *(const float2*)(xrow1 + ko);
      a0.u[v] = pk2(f0.x, f0.y);
      a1.u[v] = pk2(f1.x, f1.y);
    }
    #pragma unroll
    for (int nt = 0; nt < 4; ++nt) {
      FragU b;
      load_b(b, wdown_frag + (((nt * 16 + kb) * 32 + lane) << 3));
      acc[0][nt] = wmma_f16(a0, b, acc[0][nt]);
      acc[1][nt] = wmma_f16(a1, b, acc[1][nt]);
    }
  }

  // C/D layout: VGPR v -> M = v + 8*(lane>>4); lane&15 -> N.
  #pragma unroll
  for (int ti = 0; ti < 2; ++ti) {
    #pragma unroll
    for (int nt = 0; nt < 4; ++nt) {
      const int c = nt * 16 + m;
      const float bd = b_down[c];
      #pragma unroll
      for (int v = 0; v < 8; ++v) {
        const float g = quick_gelu(acc[ti][nt][v] + bd);
        const int row = (mt0 + ti) * 16 + v + 8 * hi;
        act[(size_t)row * DIMc + c] = (_Float16)g;
      }
    }
  }
}

// ---------------------------------------------------------------------------
// k_conv_up: fused 3x3 conv (9 taps x K=64) + channel scale + quick_gelu
//            + up-proj (K=64, N=512) + b_up -> out f32.
// Two 16-pixel M-tiles per wave.  Conv A-fragments gathered per-lane from
// shifted pixels (zero halo via predication).  Conv result staged in LDS f16
// to re-layout C -> A fragments for the up-proj WMMA.
// ---------------------------------------------------------------------------
__global__ __launch_bounds__(256)
void k_conv_up(const _Float16* __restrict__ act,
               const unsigned* __restrict__ convw_frag,
               const unsigned* __restrict__ wup_frag,
               const float* __restrict__ scale,
               const float* __restrict__ b_up,
               float* __restrict__ out) {
  __shared__ _Float16 lds[8][2][16 * DIMc];    // 4KB per wave, 32KB per block
  const int lane = threadIdx.x & 31;
  const int wave = threadIdx.x >> 5;
  const int job  = blockIdx.x * 8 + wave;      // 0..1567
  const int m = lane & 15, hi = lane >> 4;

  // per-tile geometry (tiles may straddle an image boundary: 49 tiles/img)
  int img[2], mtl[2], hh0[2], ww0[2];
  #pragma unroll
  for (int ti = 0; ti < 2; ++ti) {
    const int mt = job * 2 + ti;
    img[ti] = mt / TILES_PER_IMG;
    mtl[ti] = mt % TILES_PER_IMG;
    const int p = mtl[ti] * 16 + m;            // pixel within image (A row)
    hh0[ti] = p / Wc;
    ww0[ti] = p % Wc;
  }

  float8 yacc[2][4];
  #pragma unroll
  for (int ti = 0; ti < 2; ++ti)
    #pragma unroll
    for (int i = 0; i < 4; ++i) yacc[ti][i] = ZERO8;

  // ---- conv: 9 taps, each a 32x64 @ 64x64 GEMM fragment stream ----
  for (int t = 0; t < 9; ++t) {
    const int dh = t / 3 - 1, dw = t % 3 - 1;
    bool   ok[2];
    size_t rowoff[2];
    #pragma unroll
    for (int ti = 0; ti < 2; ++ti) {
      const int hh = hh0[ti] + dh, ww = ww0[ti] + dw;
      ok[ti] = ((unsigned)hh < (unsigned)Hc) && ((unsigned)ww < (unsigned)Wc);
      rowoff[ti] = (size_t)(img[ti] * PIX + hh * Wc + ww) * DIMc;
    }
    #pragma unroll
    for (int kb = 0; kb < 2; ++kb) {
      FragU a0, a1;
      #pragma unroll
      for (int v = 0; v < 8; ++v) {
        const int ko = kb * 32 + a_koff(v, hi);
        a0.u[v] = ok[0] ? *(const unsigned*)(act + rowoff[0] + ko) : 0u;
        a1.u[v] = ok[1] ? *(const unsigned*)(act + rowoff[1] + ko) : 0u;
      }
      #pragma unroll
      for (int nt = 0; nt < 4; ++nt) {
        FragU b;
        load_b(b, convw_frag + ((((t * 4 + nt) * 2 + kb) * 32 + lane) << 3));
        yacc[0][nt] = wmma_f16(a0, b, yacc[0][nt]);
        yacc[1][nt] = wmma_f16(a1, b, yacc[1][nt]);
      }
    }
  }

  // ---- scale + quick_gelu, stage tiles (C layout -> LDS row-major f16) ----
  #pragma unroll
  for (int ti = 0; ti < 2; ++ti) {
    _Float16* my = lds[wave][ti];
    #pragma unroll
    for (int nt = 0; nt < 4; ++nt) {
      const int c = nt * 16 + m;
      const float s = scale[c];
      #pragma unroll
      for (int v = 0; v < 8; ++v) {
        const float g = quick_gelu(yacc[ti][nt][v] * s);
        my[(v + 8 * hi) * DIMc + c] = (_Float16)g;
      }
    }
  }
  __syncthreads();

  // ---- re-gather as A fragments for up-proj (K = 64 -> 2 blocks) ----
  FragU a[2][2];                               // [tile][kb]
  #pragma unroll
  for (int ti = 0; ti < 2; ++ti) {
    const _Float16* my = lds[wave][ti];
    #pragma unroll
    for (int v = 0; v < 8; ++v) {
      a[ti][0].u[v] = *(const unsigned*)(my + m * DIMc + a_koff(v, hi));
      a[ti][1].u[v] = *(const unsigned*)(my + m * DIMc + 32 + a_koff(v, hi));
    }
  }

  const int prow0 = img[0] * PIX + mtl[0] * 16;
  const int prow1 = img[1] * PIX + mtl[1] * 16;
  for (int nt = 0; nt < 32; ++nt) {
    FragU b0, b1;
    load_b(b0, wup_frag + (((nt * 2 + 0) * 32 + lane) << 3));
    load_b(b1, wup_frag + (((nt * 2 + 1) * 32 + lane) << 3));
    float8 acc0 = ZERO8, acc1 = ZERO8;
    acc0 = wmma_f16(a[0][0], b0, acc0);
    acc0 = wmma_f16(a[0][1], b1, acc0);
    acc1 = wmma_f16(a[1][0], b0, acc1);
    acc1 = wmma_f16(a[1][1], b1, acc1);
    const int c = nt * 16 + m;
    const float bu = b_up[c];
    #pragma unroll
    for (int v = 0; v < 8; ++v) {
      out[(size_t)(prow0 + v + 8 * hi) * Cc + c] = acc0[v] + bu;
      out[(size_t)(prow1 + v + 8 * hi) * Cc + c] = acc1[v] + bu;
    }
  }
}

// ---------------------------------------------------------------------------
// Workspace layout (bytes):
//   [0,      73728)  convw_frag (18432 dwords)
//   [73728, 139264)  wdown_frag (16384 dwords)
//   [139264,204800)  wup_frag   (16384 dwords)
//   [204800, +6.4MB) act f16 [50176][64]
// Total ~6.63 MB.
// ---------------------------------------------------------------------------
extern "C" void kernel_launch(void* const* d_in, const int* in_sizes, int n_in,
                              void* d_out, int out_size, void* d_ws, size_t ws_size,
                              hipStream_t stream) {
  const float* x         = (const float*)d_in[0];
  const float* w_down    = (const float*)d_in[1];
  const float* b_down    = (const float*)d_in[2];
  const float* w_up      = (const float*)d_in[3];
  const float* b_up      = (const float*)d_in[4];
  const float* scale     = (const float*)d_in[5];
  const float* layer_emb = (const float*)d_in[6];
  const float* w_hyper   = (const float*)d_in[7];
  const float* b_hyper   = (const float*)d_in[8];
  float* out = (float*)d_out;

  char* ws = (char*)d_ws;
  unsigned*  convw_frag = (unsigned*)(ws + 0);
  unsigned*  wdown_frag = (unsigned*)(ws + 73728);
  unsigned*  wup_frag   = (unsigned*)(ws + 139264);
  _Float16*  act        = (_Float16*)(ws + 204800);

  k_prep<<<200, 256, 0, stream>>>(w_down, w_up, layer_emb, w_hyper, b_hyper,
                                  convw_frag, wdown_frag, wup_frag);
  k_down<<<NJOBS / 8, 256, 0, stream>>>(x, b_down, wdown_frag, act);
  k_conv_up<<<NJOBS / 8, 256, 0, stream>>>(act, convw_frag, wup_frag, scale,
                                           b_up, out);
}
